// RecurrentNetwork_88132728914335
// MI455X (gfx1250) — matrix operand order repaired and checked by reading, not compiled, not run on hardware
//
#include <hip/hip_runtime.h>
#include <hip/hip_bf16.h>
#include <cstdint>

// Problem constants (from the reference)
#define B_  128
#define T_  120
#define I_  4
#define C_  15
#define H_  512
#define O_  60
#define G3  (3*H_)     // 1536 stacked gates [r; z; n]

typedef __attribute__((ext_vector_type(16))) __bf16 v16bf;
typedef __attribute__((ext_vector_type(8)))  float  v8f;

// LDS row strides, padded to dodge bank conflicts while keeping 16B alignment
#define HGS  1544      // hg  [16][HGS]  f32  gate pre-activations for one M-tile
#define H32S 520       // h32 [32][H32S] f32  hidden state (exact)
#define HBS  528       // hbf [32][HBS]  bf16 hidden state (WMMA A source)

union FragBF { uint4 q[2]; v16bf v; };

// One-time fp32 -> bf16 conversion of W_hh (23.6 MB, L2-resident afterwards)
__global__ void wh_to_bf16_kernel(const float* __restrict__ wh,
                                  __bf16* __restrict__ out, int n) {
  int i = blockIdx.x * blockDim.x + threadIdx.x;
  if (i < n) out[i] = (__bf16)wh[i];
}

__device__ __forceinline__ float sigmoidf(float x) {
  return 1.0f / (1.0f + __expf(-x));
}

// Fragment loaders. Base pointers already include the lane's K sub-select.
__device__ __forceinline__ void loadA(FragBF& a, const __bf16* ra, int kb) {
  a.q[0] = *(const uint4*)(ra + kb);        // K = kb + ksel + 0..7
  a.q[1] = *(const uint4*)(ra + kb + 16);   // K = kb + 16 + ksel + 0..7
}
__device__ __forceinline__ void loadB(FragBF& b, const __bf16* rb, int kb) {
  b.q[0] = *(const uint4*)(rb + kb);        // K = kb + ksel + 0..7
  b.q[1] = *(const uint4*)(rb + kb + 16);   // K = kb + 16 + ksel + 0..7
}

#define WMMA_BF16(A, Bm, Cm) \
  __builtin_amdgcn_wmma_f32_16x16x32_bf16(false, (A).v, false, (Bm).v, \
                                          (short)0, (Cm), false, false)

// One workgroup per camera. 1024 threads = 32 waves (wave32). Each wave owns
// 3 N-tiles of 16 gate outputs (32*3*16 = 1536 = 3H). Hidden state + all
// per-camera small constants live in LDS for all 120 steps.
__global__ __launch_bounds__(1024)
void gru_cam_kernel(const float* __restrict__ inp,   // [B,T,I]
                    const int*   __restrict__ cam,   // [B]
                    const float* __restrict__ Wi,    // [C,3H,I]
                    const float* __restrict__ bi,    // [C,3H]
                    const float* __restrict__ bh,    // [C,3H]
                    const float* __restrict__ Wdec,  // [C,O,H]
                    const float* __restrict__ bdec,  // [C,O]
                    const __bf16* __restrict__ WhBf, // [C,3H,H] bf16
                    float* __restrict__ out)         // [B,O]
{
  extern __shared__ char smem[];
  float*  hg   = (float*)smem;               // [16][HGS]
  float*  h32  = hg + 16*HGS;                // [32][H32S]
  float*  wiS  = h32 + 32*H32S;              // [G3][4]  input weights (camera c)
  float*  biS  = wiS + G3*4;                 // [G3]
  float*  bhS  = biS + G3;                   // [G3]
  __bf16* hbf  = (__bf16*)(bhS + G3);        // [32][HBS]
  int*    bidx = (int*)(hbf + 32*HBS);       // [32] sample indices
  int*    cntp = bidx + 32;

  const int c   = blockIdx.x;                // camera id
  const int tid = threadIdx.x;

  // Gather this camera's samples; zero hidden state; cache per-camera consts
  if (tid == 0) {
    int n = 0;
    for (int b = 0; b < B_; ++b)
      if (cam[b] == c && n < 32) bidx[n++] = b;
    cntp[0] = n;
  }
  for (int i = tid; i < 32*H32S; i += blockDim.x) h32[i] = 0.0f;
  for (int i = tid; i < 32*HBS;  i += blockDim.x) hbf[i] = (__bf16)0.0f;
  for (int i = tid; i < G3*4;    i += blockDim.x) wiS[i] = Wi[(size_t)c*G3*4 + i];
  for (int i = tid; i < G3;      i += blockDim.x) {
    biS[i] = bi[(size_t)c*G3 + i];
    bhS[i] = bh[(size_t)c*G3 + i];
  }
  __syncthreads();

  const int count = cntp[0];
  const int nmt   = (count + 15) >> 4;       // M-tiles of 16 samples

  const int lane  = tid & 31;
  const int wv    = tid >> 5;                // wave id, owns N-tiles 3w..3w+2
  const int mrow  = lane & 15;               // A-matrix M = lane%16
  const int ncol  = lane & 15;               // B/C/D column N = lane%16
  const int kselA = (lane & 16) ? 8  : 0;    // A: lanes>=16 hold K {8..15,24..31}
  const int kselB = (lane & 16) ? 8  : 0;    // B: lanes>=16 hold K {8..15,24..31}
  const int rsel  = (lane & 16) ? 8  : 0;    // C/D: lanes>=16 hold rows v+8

  const __bf16* whc = WhBf + (size_t)c * G3 * H_;
  const int n0 = (wv*3 + 0) * 16, n1 = (wv*3 + 1) * 16, n2 = (wv*3 + 2) * 16;
  const __bf16* rb0 = whc + (size_t)(n0 + ncol)*H_ + kselB;   // B row bases
  const __bf16* rb1 = whc + (size_t)(n1 + ncol)*H_ + kselB;
  const __bf16* rb2 = whc + (size_t)(n2 + ncol)*H_ + kselB;

  for (int t = 0; t < T_; ++t) {
    for (int mt = 0; mt < nmt; ++mt) {
      const int mb = mt * 16;
      const __bf16* ra = &hbf[(mb + mrow)*HBS + kselA];       // A row base

      // ---- hg[16 x 1536] = h[16 x 512] @ Wh^T : bf16 WMMA, f32 acc.
      // Software-pipelined ping-pong: next K-chunk's fragment loads are
      // issued before the current chunk's WMMAs, hiding L2 latency.
      // `#pragma unroll 1` keeps the scheduler window to one chunk so the
      // 8 live fragments (64 VGPRs) + 24 acc VGPRs stay within the
      // 8-wave/SIMD register budget (no scratch spills).
      // EXEC is all-ones here (uniform control flow) as WMMA requires. ----
      v8f acc0 = {}, acc1 = {}, acc2 = {};
      FragBF a0, a1, b00, b10, b20, b01, b11, b21;
      loadB(b00, rb0, 0); loadB(b10, rb1, 0); loadB(b20, rb2, 0);
      loadA(a0, ra, 0);
      #pragma unroll 1
      for (int kk = 0; kk < H_/32 - 2; kk += 2) {
        const int kb0 = kk * 32, kb1 = kb0 + 32, kb2 = kb0 + 64;
        __builtin_prefetch(rb0 + kb2 + 64, 0, 1);   // global_prefetch_b8
        loadB(b01, rb0, kb1); loadB(b11, rb1, kb1); loadB(b21, rb2, kb1);
        loadA(a1, ra, kb1);
        acc0 = WMMA_BF16(a0, b00, acc0);
        acc1 = WMMA_BF16(a0, b10, acc1);
        acc2 = WMMA_BF16(a0, b20, acc2);
        loadB(b00, rb0, kb2); loadB(b10, rb1, kb2); loadB(b20, rb2, kb2);
        loadA(a0, ra, kb2);
        acc0 = WMMA_BF16(a1, b01, acc0);
        acc1 = WMMA_BF16(a1, b11, acc1);
        acc2 = WMMA_BF16(a1, b21, acc2);
      }
      // Epilogue: chunks 14 (already in a0/b*0) and 15.
      {
        const int kbL = (H_/32 - 1) * 32;           // 480
        loadB(b01, rb0, kbL); loadB(b11, rb1, kbL); loadB(b21, rb2, kbL);
        loadA(a1, ra, kbL);
        acc0 = WMMA_BF16(a0, b00, acc0);
        acc1 = WMMA_BF16(a0, b10, acc1);
        acc2 = WMMA_BF16(a0, b20, acc2);
        acc0 = WMMA_BF16(a1, b01, acc0);
        acc1 = WMMA_BF16(a1, b11, acc1);
        acc2 = WMMA_BF16(a1, b21, acc2);
      }
      #pragma unroll
      for (int v = 0; v < 8; ++v) {          // C/D layout: row = v (+8 hi lanes)
        hg[(v + rsel)*HGS + n0 + ncol] = acc0[v];
        hg[(v + rsel)*HGS + n1 + ncol] = acc1[v];
        hg[(v + rsel)*HGS + n2 + ncol] = acc2[v];
      }
      __syncthreads();

      // ---- gate nonlinearity + state update (fp32). Input projection
      // (I=4) recomputed inline from LDS-cached Wi: 12 FMAs vs 94 MB of
      // materialized x_gates traffic. ----
      for (int p = tid; p < 16*H_; p += blockDim.x) {
        const int m = p >> 9, j = p & (H_ - 1);
        const int slot = mb + m;
        if (slot < count) {
          const int b = bidx[slot];
          const float4 x = *(const float4*)&inp[((size_t)b*T_ + t)*I_];
          float xg[3];
          #pragma unroll
          for (int g = 0; g < 3; ++g) {
            const int gi = g*H_ + j;
            const float4 w = *(const float4*)&wiS[gi*4];
            xg[g] = biS[gi] + x.x*w.x + x.y*w.y + x.z*w.z + x.w*w.w;
          }
          const float hr = hg[m*HGS + j]        + bhS[j];
          const float hz = hg[m*HGS + H_ + j]   + bhS[H_ + j];
          const float hn = hg[m*HGS + 2*H_ + j] + bhS[2*H_ + j];
          const float r  = sigmoidf(xg[0] + hr);
          const float z  = sigmoidf(xg[1] + hz);
          const float nn = tanhf(xg[2] + r*hn);
          const float hp = h32[slot*H32S + j];
          const float hnew = (1.0f - z)*nn + z*hp;
          h32[slot*H32S + j] = hnew;
          hbf[slot*HBS + j]  = (__bf16)hnew;
        }
      }
      __syncthreads();
    }
  }

  // ---- per-camera decoder: out[b,:] = Wdec[c] @ h_final + bdec[c] ----
  for (int p = tid; p < count*O_; p += blockDim.x) {
    const int m = p / O_, o = p - m*O_;
    const int b = bidx[m];
    const float* wd = Wdec + ((size_t)c*O_ + o)*H_;
    float acc = bdec[c*O_ + o];
    for (int k = 0; k < H_; ++k) acc += wd[k] * h32[m*H32S + k];
    out[(size_t)b*O_ + o] = acc;
  }
}

extern "C" void kernel_launch(void* const* d_in, const int* in_sizes, int n_in,
                              void* d_out, int out_size, void* d_ws, size_t ws_size,
                              hipStream_t stream) {
  (void)in_sizes; (void)n_in; (void)out_size; (void)ws_size;
  const float* inp  = (const float*)d_in[0];
  const int*   cam  = (const int*)  d_in[1];
  const float* Wih  = (const float*)d_in[2];
  const float* Whh  = (const float*)d_in[3];
  const float* bih  = (const float*)d_in[4];
  const float* bhh  = (const float*)d_in[5];
  const float* Wdec = (const float*)d_in[6];
  const float* bdec = (const float*)d_in[7];
  float* out = (float*)d_out;

  __bf16* whbf = (__bf16*)d_ws;              // [C,3H,H] bf16, 23.6 MB (L2-resident)
  const int n = C_ * G3 * H_;
  wh_to_bf16_kernel<<<(n + 255) / 256, 256, 0, stream>>>(Whh, whbf, n);

  const size_t smem = (size_t)(16*HGS + 32*H32S + G3*4 + G3 + G3) * sizeof(float)
                    + (size_t)(32*HBS) * sizeof(__bf16)
                    + 33 * sizeof(int);      // ~236 KB of the 320 KB/WGP LDS
  (void)hipFuncSetAttribute((const void*)gru_cam_kernel,
                            hipFuncAttributeMaxDynamicSharedMemorySize, (int)smem);
  gru_cam_kernel<<<C_, 1024, smem, stream>>>(inp, cam, Wih, bih, bhh,
                                             Wdec, bdec, whbf, out);
}
